// ALiBiMultiHeadAttention_27771258536870
// MI455X (gfx1250) — compile-verified, matrix operand order checked by reading, not run on hardware
//
#include <hip/hip_runtime.h>

typedef __bf16 bf16;
typedef __attribute__((ext_vector_type(16))) __bf16 v16bf;
typedef __attribute__((ext_vector_type(8)))  float  v8f;

#define D_MODEL 1024
#define N_HEADS 16
#define D_HEAD  64
#define SEQ     2048
#define BATCH   2

__device__ __forceinline__ v8f wmma_bf16(v16bf a, v16bf b, v8f c) {
  // D = A(16x32 bf16) * B(32x16 bf16) + C(16x16 f32)
  return __builtin_amdgcn_wmma_f32_16x16x32_bf16(false, a, false, b, (short)0, c,
                                                 false, false);
}

// 16 rows x 32 contraction fragment from a row-major LDS tile [rows][ld].
// Lanes 0-15: row = row0+lane, K in {0..7, 16..23}
// Lanes 16-31: row = row0+lane-16, K in {8..15, 24..31}
// Per lane this is two contiguous 16B chunks -> two ds_load_b128.
// Requires: ld % 8 == 0 and base 16B-aligned.
__device__ __forceinline__ v16bf load_frag16x32(const bf16* base, int row0, int ld,
                                                int lane) {
  const bf16* p = base + (row0 + (lane & 15)) * ld + ((lane & 16) ? 8 : 0);
  union { uint4 u[2]; v16bf f; } t;
  t.u[0] = *(const uint4*)(p);
  t.u[1] = *(const uint4*)(p + 16);
  return t.f;
}

__global__ void cvt_f32_bf16(const float* __restrict__ in, bf16* __restrict__ out,
                             int n4) {
  const int i = blockIdx.x * blockDim.x + threadIdx.x;
  if (i < n4) {
    const float4 v = ((const float4*)in)[i];
    bf16* o = out + (size_t)i * 4;
    o[0] = (bf16)v.x; o[1] = (bf16)v.y; o[2] = (bf16)v.z; o[3] = (bf16)v.w;
  }
}

// C[M,N] = A[M,K] * W[N,K]^T + bias, bf16 inputs, fp32 accumulate.
// Block tile 128x64x32, 256 threads = 8 waves, each wave a 32x32 micro-tile.
// Global loads for the next K-tile are issued before computing the current one.
template <bool OUT_BF16>
__global__ __launch_bounds__(256) void gemm_x_wT(
    const bf16* __restrict__ A, const bf16* __restrict__ W,
    const float* __restrict__ bias, float* __restrict__ Cf, bf16* __restrict__ Cb,
    int M, int N, int K) {
  constexpr int BM = 128, BN = 64, BK = 32, LDT = BK + 8;  // padded LDS stride
  __shared__ __align__(16) bf16 As[BM * LDT];
  __shared__ __align__(16) bf16 Ws[BN * LDT];
  const int tid = threadIdx.x, lane = tid & 31, wave = tid >> 5;
  const int m0 = blockIdx.y * BM, n0 = blockIdx.x * BN;
  const int wm = (wave & 3) * 32, wn = (wave >> 2) * 32;
  const int r = tid >> 2, kc = (tid & 3) * 8;  // 8 bf16 = 16B chunks
  v8f acc[2][2] = {};

  uint4 ra0 = *(const uint4*)&A[(size_t)(m0 + r) * K + kc];
  uint4 ra1 = *(const uint4*)&A[(size_t)(m0 + r + 64) * K + kc];
  uint4 rw  = *(const uint4*)&W[(size_t)(n0 + r) * K + kc];

  for (int k0 = 0; k0 < K; k0 += BK) {
    __syncthreads();
    *(uint4*)&As[r * LDT + kc] = ra0;
    *(uint4*)&As[(r + 64) * LDT + kc] = ra1;
    *(uint4*)&Ws[r * LDT + kc] = rw;
    __syncthreads();
    if (k0 + BK < K) {  // prefetch next tile into registers
      ra0 = *(const uint4*)&A[(size_t)(m0 + r) * K + k0 + BK + kc];
      ra1 = *(const uint4*)&A[(size_t)(m0 + r + 64) * K + k0 + BK + kc];
      rw  = *(const uint4*)&W[(size_t)(n0 + r) * K + k0 + BK + kc];
    }
    const v16bf a0 = load_frag16x32(As, wm, LDT, lane);
    const v16bf a1 = load_frag16x32(As, wm + 16, LDT, lane);
    const v16bf b0 = load_frag16x32(Ws, wn, LDT, lane);
    const v16bf b1 = load_frag16x32(Ws, wn + 16, LDT, lane);
    acc[0][0] = wmma_bf16(a0, b0, acc[0][0]);
    acc[0][1] = wmma_bf16(a0, b1, acc[0][1]);
    acc[1][0] = wmma_bf16(a1, b0, acc[1][0]);
    acc[1][1] = wmma_bf16(a1, b1, acc[1][1]);
  }

  const int nl = lane & 15, mb = (lane & 16) ? 8 : 0;
#pragma unroll
  for (int mi = 0; mi < 2; ++mi)
#pragma unroll
    for (int ni = 0; ni < 2; ++ni) {
      const int n = n0 + wn + ni * 16 + nl;
      const int mrow = m0 + wm + mi * 16 + mb;
      const float bv = bias[n];
#pragma unroll
      for (int i = 0; i < 8; ++i) {
        const float v = acc[mi][ni][i] + bv;
        const size_t idx = (size_t)(mrow + i) * N + n;
        if constexpr (OUT_BF16) Cb[idx] = (bf16)v; else Cf[idx] = v;
      }
    }
}

// Flash-style ALiBi attention. Grid: (S/64 q-blocks, H, B), 128 threads (4 waves).
// qkv is (B*S, 3*D) row-major bf16; head h uses columns [h*192, h*192+192).
// V is staged TRANSPOSED in LDS (Vt[dh][key]) so P*V fragments are contiguous.
// Writes O as (B*S, D) bf16 (already the (B,S,H*dh) layout the out-proj wants).
__global__ __launch_bounds__(128) void attn_alibi(const bf16* __restrict__ qkv,
                                                  bf16* __restrict__ obuf) {
  constexpr int D3 = 3 * D_MODEL;
  constexpr int LD = 72;  // padded stride (multiple of 8 -> keeps 16B alignment)
  __shared__ __align__(16) bf16 Qs[64 * LD];
  __shared__ __align__(16) bf16 Ks[64 * LD];
  __shared__ __align__(16) bf16 Vt[64 * LD];  // [dh][key]
  __shared__ __align__(16) bf16 Ps[64 * LD];
  __shared__ float Ss[64 * 65];               // padded fp32 scores
  __shared__ float alpha_s[64];
  __shared__ float l_s[64];

  const int tid = threadIdx.x, lane = tid & 31, wave = tid >> 5;
  const int qb = blockIdx.x, h = blockIdx.y, b = blockIdx.z;
  const float slope = exp2f(-(float)h);  // 0.5^h
  const float scale = 0.125f;            // 1/sqrt(64)
  const size_t rowbase = (size_t)b * SEQ * D3;
  const int colq = h * (3 * D_HEAD);
  const int colk = colq + D_HEAD;
  const int colv = colq + 2 * D_HEAD;
  const int nl = lane & 15, mb = (lane & 16) ? 8 : 0;
  const int r = tid >> 1, hh = (tid & 1) * 32;  // staging coords

  {  // stage Q block (64 x 64)
    const bf16* src = qkv + rowbase + (size_t)(qb * 64 + r) * D3 + colq + hh;
#pragma unroll
    for (int c = 0; c < 4; ++c)
      *(uint4*)&Qs[r * LD + hh + c * 8] = *(const uint4*)&src[c * 8];
  }
  __syncthreads();
  // Q fragments are loop-invariant: load once.
  const v16bf qa0 = load_frag16x32(Qs, wave * 16, LD, lane);       // dh 0..31
  const v16bf qa1 = load_frag16x32(Qs + 32, wave * 16, LD, lane);  // dh 32..63

  v8f o[4] = {};
  float mrow = -3.0e38f, lrow = 0.0f;

  for (int j = 0; j < SEQ / 64; ++j) {
    __syncthreads();  // previous P*V done before K/V overwrite
    {
      const bf16* ksrc = qkv + rowbase + (size_t)(j * 64 + r) * D3 + colk + hh;
      const bf16* vsrc = qkv + rowbase + (size_t)(j * 64 + r) * D3 + colv + hh;
      uint4 kreg[4], vreg[4];
#pragma unroll
      for (int c = 0; c < 4; ++c) {
        kreg[c] = *(const uint4*)&ksrc[c * 8];
        vreg[c] = *(const uint4*)&vsrc[c * 8];
      }
      if (j + 1 < SEQ / 64) {  // global_prefetch_b8 the next KV block
        __builtin_prefetch(ksrc + (size_t)64 * D3, 0, 1);
        __builtin_prefetch(vsrc + (size_t)64 * D3, 0, 1);
      }
#pragma unroll
      for (int c = 0; c < 4; ++c)
        *(uint4*)&Ks[r * LD + hh + c * 8] = kreg[c];
      // V transposed scatter: Vt[dh][key]
#pragma unroll
      for (int c = 0; c < 4; ++c) {
        union { uint4 u; bf16 e[8]; } tv;
        tv.u = vreg[c];
#pragma unroll
        for (int e = 0; e < 8; ++e) Vt[(hh + c * 8 + e) * LD + r] = tv.e[e];
      }
    }
    __syncthreads();

    // S = Q K^T : each wave 16 q-rows x 64 keys (8 WMMAs)
    v8f s[4] = {};
#pragma unroll
    for (int t = 0; t < 4; ++t) {
      const v16bf kb0 = load_frag16x32(Ks, t * 16, LD, lane);
      const v16bf kb1 = load_frag16x32(Ks + 32, t * 16, LD, lane);
      s[t] = wmma_bf16(qa0, kb0, s[t]);
      s[t] = wmma_bf16(qa1, kb1, s[t]);
    }
    // fuse 1/sqrt(dh) scale + ALiBi bias, park fp32 scores in LDS
#pragma unroll
    for (int t = 0; t < 4; ++t) {
      const int col = j * 64 + t * 16 + nl;
#pragma unroll
      for (int i = 0; i < 8; ++i) {
        const int rloc = wave * 16 + mb + i;
        const int row = qb * 64 + rloc;
        Ss[rloc * 65 + t * 16 + nl] =
            s[t][i] * scale - fabsf((float)(row - col)) * slope;
      }
    }
    __syncthreads();

    // online softmax: thread r owns q-row r
    if (tid < 64) {
      const float* sr = &Ss[tid * 65];
      float mx = -3.0e38f;
#pragma unroll 8
      for (int c = 0; c < 64; ++c) mx = fmaxf(mx, sr[c]);
      const float newm = fmaxf(mrow, mx);
      const float al = __expf(mrow - newm);
      float sum = 0.0f;
#pragma unroll 8
      for (int c = 0; c < 64; ++c) {
        const float p = __expf(sr[c] - newm);
        sum += p;
        Ps[tid * LD + c] = (bf16)p;
      }
      lrow = lrow * al + sum;
      mrow = newm;
      alpha_s[tid] = al;
    }
    __syncthreads();

    // O = O*alpha + P V (8 WMMAs per wave)
    float alv[8];
#pragma unroll
    for (int i = 0; i < 8; ++i) alv[i] = alpha_s[wave * 16 + mb + i];
    const v16bf p0 = load_frag16x32(Ps, wave * 16, LD, lane);       // keys 0..31
    const v16bf p1 = load_frag16x32(Ps + 32, wave * 16, LD, lane);  // keys 32..63
#pragma unroll
    for (int t = 0; t < 4; ++t) {
#pragma unroll
      for (int i = 0; i < 8; ++i) o[t][i] *= alv[i];
      const v16bf v0 = load_frag16x32(Vt, t * 16, LD, lane);        // keys 0..31
      const v16bf v1 = load_frag16x32(Vt + 32, t * 16, LD, lane);   // keys 32..63
      o[t] = wmma_bf16(p0, v0, o[t]);
      o[t] = wmma_bf16(p1, v1, o[t]);
    }
  }

  if (tid < 64) l_s[tid] = lrow;
  __syncthreads();

#pragma unroll
  for (int t = 0; t < 4; ++t)
#pragma unroll
    for (int i = 0; i < 8; ++i) {
      const int rloc = wave * 16 + mb + i;
      const float v = o[t][i] / l_s[rloc];
      const size_t srow = (size_t)b * SEQ + qb * 64 + rloc;
      obuf[srow * D_MODEL + h * D_HEAD + t * 16 + nl] = (bf16)v;
    }
}

extern "C" void kernel_launch(void* const* d_in, const int* in_sizes, int n_in,
                              void* d_out, int out_size, void* d_ws, size_t ws_size,
                              hipStream_t stream) {
  (void)in_sizes; (void)n_in; (void)out_size; (void)ws_size;
  const float* x     = (const float*)d_in[0];
  const float* qkv_w = (const float*)d_in[1];
  const float* qkv_b = (const float*)d_in[2];
  const float* out_w = (const float*)d_in[3];
  const float* out_b = (const float*)d_in[4];
  float* out = (float*)d_out;

  const size_t M = (size_t)BATCH * SEQ;  // 4096
  bf16* x_bf    = (bf16*)d_ws;                                   // 4096x1024
  bf16* qkvw_bf = x_bf + M * D_MODEL;                            // 3072x1024
  bf16* outw_bf = qkvw_bf + (size_t)3 * D_MODEL * D_MODEL;       // 1024x1024
  bf16* qkv_bf  = outw_bf + (size_t)D_MODEL * D_MODEL;           // 4096x3072
  bf16* o_bf    = qkv_bf + M * 3 * D_MODEL;                      // 4096x1024
  // total workspace: ~48 MB

  const int T = 256;
  int n4 = (int)(M * D_MODEL) / 4;
  cvt_f32_bf16<<<(n4 + T - 1) / T, T, 0, stream>>>(x, x_bf, n4);
  n4 = 3 * D_MODEL * D_MODEL / 4;
  cvt_f32_bf16<<<(n4 + T - 1) / T, T, 0, stream>>>(qkv_w, qkvw_bf, n4);
  n4 = D_MODEL * D_MODEL / 4;
  cvt_f32_bf16<<<(n4 + T - 1) / T, T, 0, stream>>>(out_w, outw_bf, n4);

  // QKV projection: (4096x1024) @ (3072x1024)^T -> bf16 (4096x3072)
  gemm_x_wT<true><<<dim3(3 * D_MODEL / 64, (int)M / 128), 256, 0, stream>>>(
      x_bf, qkvw_bf, qkv_b, nullptr, qkv_bf, (int)M, 3 * D_MODEL, D_MODEL);

  // flash attention with ALiBi
  attn_alibi<<<dim3(SEQ / 64, N_HEADS, BATCH), 128, 0, stream>>>(qkv_bf, o_bf);

  // output projection: (4096x1024) @ (1024x1024)^T -> fp32 d_out
  gemm_x_wT<false><<<dim3(D_MODEL / 64, (int)M / 128), 256, 0, stream>>>(
      o_bf, outw_bf, out_b, out, nullptr, (int)M, D_MODEL, D_MODEL);
}
